// LAMME_5136780886252
// MI455X (gfx1250) — matrix operand order repaired and checked by reading, not compiled, not run on hardware
//
#include <hip/hip_runtime.h>
#include <hip/hip_bf16.h>

// ---------------------------------------------------------------------------
// LAM+ME block for MI455X (gfx1250, wave32, WMMA).
// Pipeline:
//   K0 prep_weights : net_w f32 (co,ci,3,3) -> wB bf16 [tap][co][ci]
//   K1 conv_wmma    : 3x3 conv via implicit GEMM, v_wmma_f32_16x16x32_bf16,
//                     B tiles staged with GLOBAL_LOAD_ASYNC_TO_LDS_B128
//   K2 kpool        : spatial mean m[nt][c] of conv_out
//   K3 kxg          : x_g = mean_t(m) @ lam_w.T + lam_b
//   K4 kwgt         : temporal MLP + BN + softmax -> wgt[n][c][3]
//   K5 klam         : depthwise temporal conv -> lam_out (into d_out)
//                     + spatial means s_lam[nt][c]
//   K6 kgate        : ME frame-diff means + 3-tap channel conv + sigmoid
//   K7 kmul         : d_out *= gate  (in place)
// ---------------------------------------------------------------------------

typedef __attribute__((ext_vector_type(16))) __bf16 v16bf;
typedef __attribute__((ext_vector_type(8)))  float  v8f;
typedef int gv4i __attribute__((vector_size(16)));   // b128 carrier for async DMA

#define N_IMG 8
#define T_SEG 8
#define NT    64            // N_IMG * T_SEG
#define C_CH  256
#define HDIM  56
#define HW    3136          // 56*56
#define KSZ   3

#if defined(__has_builtin)
#if __has_builtin(__builtin_amdgcn_global_load_async_to_lds_b128)
#define USE_ASYNC_LDS 1
#endif
#endif

struct U32x2 { uint4 a, b; };   // 32B carrier for v16bf bit_cast

__device__ __forceinline__ __bf16 f2bf(float f) {
    union { float f; unsigned int u; } v; v.f = f;
    unsigned int r = v.u + 0x7FFFu + ((v.u >> 16) & 1u);   // RNE
    unsigned short h = (unsigned short)(r >> 16);
    return __builtin_bit_cast(__bf16, h);
}

__device__ __forceinline__ float blockReduceSum(float v, float* sdata) {
    int tid = threadIdx.x;
    sdata[tid] = v;
    __syncthreads();
    for (int s = blockDim.x >> 1; s > 0; s >>= 1) {
        if (tid < s) sdata[tid] += sdata[tid + s];
        __syncthreads();
    }
    float r = sdata[0];
    __syncthreads();
    return r;
}

// ---------------------------------------------------------------------------
// K0: net_w (co,ci,3,3) f32 -> wB[tap][co][ci] bf16  (ci contiguous)
// ---------------------------------------------------------------------------
__global__ __launch_bounds__(256) void prep_weights(
        const float* __restrict__ net_w, __bf16* __restrict__ wB) {
    int o = blockIdx.x * 256 + threadIdx.x;          // 9*256*256 = 589824
    if (o >= 9 * C_CH * C_CH) return;
    int tap = o / (C_CH * C_CH);
    int rem = o - tap * (C_CH * C_CH);
    int co  = rem >> 8;
    int ci  = rem & 255;
    wB[o] = f2bf(net_w[(co * C_CH + ci) * 9 + tap]);
}

// ---------------------------------------------------------------------------
// K1: implicit-GEMM 3x3 conv, bf16 WMMA, f32 accumulate.
// grid = (56 rows, 64 images, 2 co-blocks), block = 256 (8 waves).
// Each wave: M=16 pixels x N=64 couts (4 WMMA tiles), K walked 32 at a time.
// LDS: xs[row3][col64][ci32] (K contiguous), bs3[kw3][co128][ci32].
// B tiles are pure byte-copies from wB[tap][co][ci] -> async DMA to LDS.
// ---------------------------------------------------------------------------
__global__ __launch_bounds__(256) void conv_wmma(
        const float*  __restrict__ x,     // (nt, c, 56, 56)
        const __bf16* __restrict__ wB,    // [tap][co][ci]
        const float*  __restrict__ bias,  // (c,)
        float*        __restrict__ out)   // (nt, c, 56, 56)
{
    __shared__ __bf16 xs[3 * 64 * 32];      // 12 KB
    __shared__ __bf16 bs3[3 * 128 * 32];    // 24 KB

    const int h    = blockIdx.x;            // output row
    const int b    = blockIdx.y;            // nt index
    const int cob  = blockIdx.z * 128;      // cout block base
    const int tid  = threadIdx.x;
    const int lane = tid & 31;
    const int wid  = tid >> 5;
    const int wave_m = wid & 3;             // 0..3  -> pixel tile
    const int wave_n = wid >> 2;            // 0..1  -> cout 64-tile
    const int half = lane >> 4;
    const int l16  = lane & 15;

    v8f acc[4] = {};                        // 4 x (16x16 f32)

    for (int chunk = 0; chunk < 8; ++chunk) {
        const int cib = chunk * 32;
        for (int kh = 0; kh < 3; ++kh) {
            if (kh == 0) {
                // stage x slab: rows h-1..h+1, cols -1..56 (padded to 64), 32 ci
                for (int e = tid; e < 3 * 64 * 32; e += 256) {
                    int col  = e & 63;
                    int ci_l = (e >> 6) & 31;
                    int row  = e >> 11;
                    int hh = h + row - 1;
                    int ww = col - 1;
                    float v = 0.0f;
                    if (hh >= 0 && hh < HDIM && ww >= 0 && ww < HDIM) {
                        const float* src =
                            &x[(((size_t)b * C_CH + (cib + ci_l)) * HDIM + hh) * HDIM + ww];
                        v = *src;
                        if (chunk < 7)                         // warm next K-chunk
                            __builtin_prefetch(src + (size_t)32 * HW, 0, 3);
                    }
                    xs[(row * 64 + col) * 32 + ci_l] = f2bf(v);
                }
            }
            // stage B tiles for taps (kh,0..2): 128 co x 32 ci each, 64B rows.
            // 1536 x 16-byte transfers, 6 per thread.
            for (int e = tid; e < 1536; e += 256) {
                int seg  = e & 3;                 // 8-ci segment within row
                int co_l = (e >> 2) & 127;
                int kw   = e >> 9;
                const __bf16* src =
                    &wB[(((size_t)(kh * 3 + kw)) * C_CH + (cob + co_l)) * C_CH
                        + cib + seg * 8];
                __bf16* dst = &bs3[(kw * 128 + co_l) * 32 + seg * 8];
#ifdef USE_ASYNC_LDS
                __builtin_amdgcn_global_load_async_to_lds_b128(
                    (__attribute__((address_space(1))) gv4i*)const_cast<__bf16*>(src),
                    (__attribute__((address_space(3))) gv4i*)dst, 0, 0);
#else
                *(uint4*)dst = *(const uint4*)src;
#endif
            }
#ifdef USE_ASYNC_LDS
#if __has_builtin(__builtin_amdgcn_s_wait_asynccnt)
            __builtin_amdgcn_s_wait_asynccnt(0);
#else
            asm volatile("s_wait_asynccnt 0x0" ::: "memory");
#endif
#endif
            __syncthreads();

            #pragma unroll
            for (int kw = 0; kw < 3; ++kw) {
                // A fragment (16x32 bf16): lane l16 = pixel M, per ISA layout
                const int p = wave_m * 16 + l16;
                const int abase = (kh * 64 + (p + kw)) * 32;
                uint4 lo = *(const uint4*)&xs[abase + half * 16];       // K[half*16 .. +7]
                uint4 hi = *(const uint4*)&xs[abase + half * 16 + 8];   // K[half*16+8 ..]
                U32x2 cat; cat.a = lo; cat.b = hi;
                v16bf afrag = __builtin_bit_cast(v16bf, cat);

                #pragma unroll
                for (int nt_ = 0; nt_ < 4; ++nt_) {
                    const int co_l = wave_n * 64 + nt_ * 16 + l16;      // N = l16 tile
                    const v16bf bfrag =
                        *(const v16bf*)&bs3[(kw * 128 + co_l) * 32 + half * 16];
                    acc[nt_] = __builtin_amdgcn_wmma_f32_16x16x32_bf16(
                        false, afrag, false, bfrag, (short)0, acc[nt_], false, false);
                }
            }
            __syncthreads();
        }
    }

    // epilogue: + bias, store (D layout: vgpr r -> M=r+half*8, N=l16)
    #pragma unroll
    for (int nt_ = 0; nt_ < 4; ++nt_) {
        union { v8f v; float f[8]; } u; u.v = acc[nt_];
        const int co = cob + wave_n * 64 + nt_ * 16 + l16;
        const float bv = bias[co];
        #pragma unroll
        for (int r = 0; r < 8; ++r) {
            int wpix = wave_m * 16 + r + half * 8;
            if (wpix < HDIM)
                out[(((size_t)b * C_CH + co) * HDIM + h) * HDIM + wpix] = u.f[r] + bv;
        }
    }
}

// ---------------------------------------------------------------------------
// K2: m[nt][c] = mean_hw(conv_out[nt][c]); grid (c, nt), 128 threads
// ---------------------------------------------------------------------------
__global__ __launch_bounds__(128) void kpool(
        const float* __restrict__ conv_out, float* __restrict__ m) {
    __shared__ float sdata[128];
    int c  = blockIdx.x, nt = blockIdx.y;
    const float* row = &conv_out[((size_t)nt * C_CH + c) * HW];
    float s = 0.f;
    for (int i = threadIdx.x; i < HW; i += 128) s += row[i];
    s = blockReduceSum(s, sdata);
    if (threadIdx.x == 0) m[nt * C_CH + c] = s * (1.0f / HW);
}

// ---------------------------------------------------------------------------
// K3: x_g[n][co] = (mean_t m) @ lam_w.T + lam_b ; grid n=8, 256 threads
// ---------------------------------------------------------------------------
__global__ __launch_bounds__(256) void kxg(
        const float* __restrict__ m, const float* __restrict__ lam_w,
        const float* __restrict__ lam_b, float* __restrict__ xg) {
    __shared__ float s1[C_CH];
    int n = blockIdx.x, c = threadIdx.x;
    float s = 0.f;
    for (int t = 0; t < T_SEG; ++t) s += m[(n * T_SEG + t) * C_CH + c];
    s1[c] = s * (1.0f / T_SEG);
    __syncthreads();
    float acc = lam_b[c];
    for (int ci = 0; ci < C_CH; ++ci) acc += s1[ci] * lam_w[c * C_CH + ci];
    xg[n * C_CH + c] = acc;
}

// ---------------------------------------------------------------------------
// K4: temporal MLP + BN + ReLU + softmax -> wgt[n*c][3]; 8 blocks x 256
// ---------------------------------------------------------------------------
__global__ __launch_bounds__(256) void kwgt(
        const float* __restrict__ m, const float* __restrict__ xg,
        const float* __restrict__ w1, const float* __restrict__ gma,
        const float* __restrict__ bta, const float* __restrict__ mu,
        const float* __restrict__ var, const float* __restrict__ w2,
        float* __restrict__ wgt) {
    int idx = blockIdx.x * 256 + threadIdx.x;     // (n,c) flat, 2048 total
    int n = idx >> 8, c = idx & 255;
    float pooled[T_SEG];
    float xgv = xg[n * C_CH + c];
    for (int t = 0; t < T_SEG; ++t)
        pooled[t] = m[(n * T_SEG + t) * C_CH + c] + xgv;
    float hdn[2 * T_SEG];
    for (int j = 0; j < 2 * T_SEG; ++j) {
        float a = 0.f;
        for (int t = 0; t < T_SEG; ++t) a += pooled[t] * w1[j * T_SEG + t];
        float sc = gma[j] * rsqrtf(var[j] + 1e-5f);
        a = (a - mu[j]) * sc + bta[j];
        hdn[j] = a > 0.f ? a : 0.f;
    }
    float lg[KSZ];
    float mx = -3.4e38f;
    for (int k = 0; k < KSZ; ++k) {
        float a = 0.f;
        for (int j = 0; j < 2 * T_SEG; ++j) a += hdn[j] * w2[k * 2 * T_SEG + j];
        lg[k] = a; mx = fmaxf(mx, a);
    }
    float den = 0.f;
    for (int k = 0; k < KSZ; ++k) { lg[k] = expf(lg[k] - mx); den += lg[k]; }
    for (int k = 0; k < KSZ; ++k) wgt[idx * KSZ + k] = lg[k] / den;
}

// ---------------------------------------------------------------------------
// K5: lam_out (-> d_out) + spatial means s_lam; grid (nt, c), 256 threads
// ---------------------------------------------------------------------------
__global__ __launch_bounds__(256) void klam(
        const float* __restrict__ conv_out, const float* __restrict__ xg,
        const float* __restrict__ wgt, float* __restrict__ lam_out,
        float* __restrict__ s_lam) {
    __shared__ float sdata[256];
    int nt = blockIdx.x, c = blockIdx.y;
    int n = nt >> 3, tloc = nt & 7;
    float w0 = wgt[(n * C_CH + c) * KSZ + 0];
    float w1 = wgt[(n * C_CH + c) * KSZ + 1];
    float w2 = wgt[(n * C_CH + c) * KSZ + 2];
    float xgv = xg[n * C_CH + c];
    const float* base = &conv_out[((size_t)n * T_SEG) * C_CH * HW + (size_t)c * HW];
    float* dst = &lam_out[((size_t)nt * C_CH + c) * HW];
    float lsum = 0.f;
    for (int i = threadIdx.x; i < HW; i += 256) {
        float acc = 0.f;
        int tm = tloc - 1, t0 = tloc, tp = tloc + 1;
        if (tm >= 0)    acc += w0 * (base[(size_t)tm * C_CH * HW + i] + xgv);
                        acc += w1 * (base[(size_t)t0 * C_CH * HW + i] + xgv);
        if (tp < T_SEG) acc += w2 * (base[(size_t)tp * C_CH * HW + i] + xgv);
        dst[i] = acc;
        lsum += acc;
    }
    lsum = blockReduceSum(lsum, sdata);
    if (threadIdx.x == 0) s_lam[nt * C_CH + c] = lsum * (1.0f / HW);
}

// ---------------------------------------------------------------------------
// K6: ME gate; grid nt=64, 256 threads (one per channel)
// ---------------------------------------------------------------------------
__global__ __launch_bounds__(256) void kgate(
        const float* __restrict__ s_lam, const float* __restrict__ me_w,
        float* __restrict__ gate) {
    int nt = blockIdx.x, c = threadIdx.x;
    int tloc = nt & 7;
    auto yval = [&](int cc) -> float {
        if (cc < 0 || cc >= C_CH || tloc >= T_SEG - 1) return 0.f;
        return s_lam[(nt + 1) * C_CH + cc] - s_lam[nt * C_CH + cc];
    };
    float yc = me_w[0] * yval(c - 1) + me_w[1] * yval(c) + me_w[2] * yval(c + 1);
    gate[nt * C_CH + c] = 1.f / (1.f + expf(-yc));
}

// ---------------------------------------------------------------------------
// K7: d_out *= gate (broadcast over hw); grid nt*c = 16384, 256 threads
// ---------------------------------------------------------------------------
__global__ __launch_bounds__(256) void kmul(
        float* __restrict__ out, const float* __restrict__ gate) {
    float g = gate[blockIdx.x];
    float* row = &out[(size_t)blockIdx.x * HW];
    for (int i = threadIdx.x; i < HW; i += 256) row[i] *= g;
}

// ---------------------------------------------------------------------------
extern "C" void kernel_launch(void* const* d_in, const int* in_sizes, int n_in,
                              void* d_out, int out_size, void* d_ws, size_t ws_size,
                              hipStream_t stream) {
    (void)in_sizes; (void)n_in; (void)out_size; (void)ws_size;

    const float* x      = (const float*)d_in[0];
    const float* net_w  = (const float*)d_in[1];
    const float* net_b  = (const float*)d_in[2];
    const float* lam_w  = (const float*)d_in[3];
    const float* lam_b  = (const float*)d_in[4];
    const float* mlp_w1 = (const float*)d_in[5];
    const float* bn_g   = (const float*)d_in[6];
    const float* bn_b   = (const float*)d_in[7];
    const float* bn_m   = (const float*)d_in[8];
    const float* bn_v   = (const float*)d_in[9];
    const float* mlp_w2 = (const float*)d_in[10];
    const float* me_w   = (const float*)d_in[11];
    float* out = (float*)d_out;

    char* ws = (char*)d_ws;
    float*  conv_out = (float*)(ws);                       // 205,520,896 B
    __bf16* wB       = (__bf16*)(ws + 205520896);          //   1,179,648 B
    float*  m        = (float*)(ws + 206700544);           //      65,536 B
    float*  xg       = (float*)(ws + 206766080);           //       8,192 B
    float*  wgt      = (float*)(ws + 206774272);           //      24,576 B
    float*  s_lam    = (float*)(ws + 206798848);           //      65,536 B
    float*  gate     = (float*)(ws + 206864384);           //      65,536 B

    prep_weights<<<dim3((9 * C_CH * C_CH + 255) / 256), dim3(256), 0, stream>>>(net_w, wB);
    conv_wmma<<<dim3(HDIM, NT, 2), dim3(256), 0, stream>>>(x, wB, net_b, conv_out);
    kpool<<<dim3(C_CH, NT), dim3(128), 0, stream>>>(conv_out, m);
    kxg<<<dim3(N_IMG), dim3(256), 0, stream>>>(m, lam_w, lam_b, xg);
    kwgt<<<dim3(8), dim3(256), 0, stream>>>(m, xg, mlp_w1, bn_g, bn_b, bn_m, bn_v, mlp_w2, wgt);
    klam<<<dim3(NT, C_CH), dim3(256), 0, stream>>>(conv_out, xg, wgt, out, s_lam);
    kgate<<<dim3(NT), dim3(256), 0, stream>>>(s_lam, me_w, gate);
    kmul<<<dim3(NT * C_CH), dim3(256), 0, stream>>>(out, gate);
}
